// ThalamicNeuralODE_24850680775005
// MI455X (gfx1250) — matrix operand order repaired and checked by reading, not compiled, not run on hardware
//
#include <hip/hip_runtime.h>
#include <hip/hip_bf16.h>

// ---------------------------------------------------------------------------
// Thalamic Neural ODE (RK4 scan) for gfx1250 / MI455X.
// One persistent workgroup per 16-row batch tile (512/16 = 32 blocks).
// Weights live in LDS as bf16, transposed [N][Kpad] so WMMA B-fragments are
// two 16B LDS reads per lane. All matmuls -> v_wmma_f32_16x16x32_bf16.
// RK4 state z kept in fp32 *registers* of the owning waves; activations use
// branch-free hardware transcendentals (v_tanh_f32 / v_exp_f32 / v_rcp_f32).
// ---------------------------------------------------------------------------

typedef __bf16 bf16;
typedef __attribute__((ext_vector_type(16))) __bf16 v16bf;
typedef __attribute__((ext_vector_type(8)))  __bf16 v8bf;
typedef __attribute__((ext_vector_type(8)))  float  v8f;

#define LATENT   64
#define INPUT    21
#define HIDDEN   128
#define OUTPUT   20
#define SEQ      2048
#define BATCH    512
#define DT       1.0f
#define NTHREADS 256   // 8 waves of 32

// ---- LDS layout (bytes) ----------------------------------------------------
#define OFF_WTE1   0                         // [128][32]  bf16 :  8192
#define OFF_WTE2   (OFF_WTE1 + 128*32*2)     // [64][128]  bf16 : 16384
#define OFF_WT1    (OFF_WTE2 + 64*128*2)     // [128][96]  bf16 : 24576
#define OFF_WT2    (OFF_WT1  + 128*96*2)     // [128][128] bf16 : 32768
#define OFF_WT3    (OFF_WT2  + 128*128*2)    // [64][128]  bf16 : 16384
#define OFF_WTD1   (OFF_WT3  + 64*128*2)     // [128][64]  bf16 : 16384
#define OFF_WTD2   (OFF_WTD1 + 128*64*2)     // [32][128]  bf16 :  8192
#define OFF_BE1    (OFF_WTD2 + 32*128*2)     // f32[128]
#define OFF_BE2    (OFF_BE1 + 128*4)         // f32[64]
#define OFF_B1     (OFF_BE2 + 64*4)          // f32[128]
#define OFF_B2     (OFF_B1  + 128*4)         // f32[128]
#define OFF_B3     (OFF_B2  + 128*4)         // f32[64]
#define OFF_BD1    (OFF_B3  + 64*4)          // f32[128]
#define OFF_BD2    (OFF_BD1 + 128*4)         // f32[32]
#define OFF_ZB     (OFF_BD2 + 32*4)          // bf16[16][64]
#define OFF_ZPERT  (OFF_ZB   + 16*64*2)      // bf16[16][64]
#define OFF_UBUF   (OFF_ZPERT+ 16*64*2)      // bf16[16][32]
#define OFF_HA     (OFF_UBUF + 16*32*2)      // bf16[16][128]
#define OFF_HB     (OFF_HA   + 16*128*2)     // bf16[16][128]
#define SMEM_BYTES (OFF_HB   + 16*128*2)     // ~139 KB

// ---- branch-free hardware transcendentals ----------------------------------
#define LOG2E  1.4426950408889634f

__device__ __forceinline__ float fast_tanh(float x) {
#if defined(__HIP_DEVICE_COMPILE__)
#if __has_builtin(__builtin_amdgcn_tanhf)
  return __builtin_amdgcn_tanhf(x);            // v_tanh_f32
#else
  // tanh(x) = 1 - 2/(1 + 2^(2x*log2e)); saturates correctly via exp2 inf/0.
  const float e = __builtin_amdgcn_exp2f(x * (2.0f * LOG2E));
  return 1.0f - 2.0f * __builtin_amdgcn_rcpf(1.0f + e);
#endif
#else
  return x;  // host-pass stub; never executed
#endif
}

__device__ __forceinline__ float fast_sigmoid(float x) {
#if defined(__HIP_DEVICE_COMPILE__)
  const float e = __builtin_amdgcn_exp2f(-x * LOG2E);   // v_exp_f32
  return __builtin_amdgcn_rcpf(1.0f + e);               // v_rcp_f32
#else
  return x;  // host-pass stub; never executed
#endif
}

// ---- fragment helpers ------------------------------------------------------
__device__ __forceinline__ v8f vzero() {
  v8f z;
#pragma unroll
  for (int i = 0; i < 8; ++i) z[i] = 0.0f;
  return z;
}

// A-fragment: 16(M) x 32(K) bf16 from row-major LDS [16][ld], K base kb.
__device__ __forceinline__ v16bf load_a32(const bf16* base, int ld, int kb, int lane) {
  const int m  = lane & 15;
  const int kh = (lane >> 4) << 3;                 // 0 or 8
  const bf16* p = base + m * ld + kb + kh;
  v8bf lo = *(const v8bf*)p;                       // K = kb+kh   .. +7
  v8bf hi = *(const v8bf*)(p + 16);                // K = kb+16+kh.. +7
  return __builtin_shufflevector(lo, hi, 0,1,2,3,4,5,6,7,8,9,10,11,12,13,14,15);
}

// B-fragment: 32(K) x 16(N) from transposed weights Wt[N][ldk] (bf16).
__device__ __forceinline__ v16bf load_b32(const bf16* wt, int ldk, int nb, int kb, int lane) {
  const int n  = nb + (lane & 15);
  const int kh = (lane >> 4) << 3;                 // 0 or 8
  const bf16* p = wt + n * ldk + kb + kh;
  v8bf lo = *(const v8bf*)p;
  v8bf hi = *(const v8bf*)(p + 16);
  return __builtin_shufflevector(lo, hi, 0,1,2,3,4,5,6,7,8,9,10,11,12,13,14,15);
}

__device__ __forceinline__ v8f wmma_bf16(v16bf a, v16bf b, v8f c) {
  return __builtin_amdgcn_wmma_f32_16x16x32_bf16(false, a, false, b,
                                                 (short)0, c, false, false);
}

// Write one 16x16 C tile as tanh(acc+bias) in bf16, row-major LDS.
__device__ __forceinline__ void store_act_tanh(bf16* dst, int ld, int nb, int lane,
                                               v8f acc, const float* bias) {
  const int n  = nb + (lane & 15);
  const int mb = (lane >> 4) << 3;
  const float bn = bias[n];
#pragma unroll
  for (int r = 0; r < 8; ++r)
    dst[(mb + r) * ld + n] = (bf16)fast_tanh(acc[r] + bn);
}

// ---- ODE vector field f([z,u]) ---------------------------------------------
// 8 waves: L1/L2 one 16-col tile per wave; L3 (64 cols) on waves 0..3.
__device__ __forceinline__ v8f ode_f(const bf16* zsrc, const bf16* ubuf,
                                     const bf16* Wt1, const float* b1,
                                     const bf16* Wt2, const float* b2,
                                     const bf16* Wt3, const float* b3,
                                     bf16* hA, bf16* hB, int w, int lane) {
  {
    const int nb = w * 16;
    v8f acc = vzero();
    acc = wmma_bf16(load_a32(zsrc, 64, 0,  lane), load_b32(Wt1, 96, nb, 0,  lane), acc);
    acc = wmma_bf16(load_a32(zsrc, 64, 32, lane), load_b32(Wt1, 96, nb, 32, lane), acc);
    acc = wmma_bf16(load_a32(ubuf, 32, 0,  lane), load_b32(Wt1, 96, nb, 64, lane), acc);
    store_act_tanh(hA, HIDDEN, nb, lane, acc, b1);
  }
  __syncthreads();
  {
    const int nb = w * 16;
    v8f acc = vzero();
#pragma unroll
    for (int kb = 0; kb < 128; kb += 32)
      acc = wmma_bf16(load_a32(hA, HIDDEN, kb, lane), load_b32(Wt2, 128, nb, kb, lane), acc);
    store_act_tanh(hB, HIDDEN, nb, lane, acc, b2);
  }
  __syncthreads();
  v8f acc = vzero();
  if (w < 4) {
    const int nb = w * 16;
#pragma unroll
    for (int kb = 0; kb < 128; kb += 32)
      acc = wmma_bf16(load_a32(hB, HIDDEN, kb, lane), load_b32(Wt3, 128, nb, kb, lane), acc);
    const float bn = b3[nb + (lane & 15)];
#pragma unroll
    for (int r = 0; r < 8; ++r) acc[r] += bn;
  }
  return acc;
}

// ---- weight / bias preload (f32 global -> bf16 transposed LDS) -------------
__device__ void load_wt(const float* w, int Kreal, int Nmem, bf16* Wt,
                        int Kpad, int Npad, int tid) {
  const int total = Npad * Kpad;
  for (int i = tid; i < total; i += NTHREADS) {
    const int n = i / Kpad, k = i % Kpad;
    const float v = (k < Kreal && n < Nmem) ? w[k * Nmem + n] : 0.0f;
    Wt[i] = (bf16)v;
  }
}
__device__ void load_bias(const float* b, int Nreal, float* dst, int Npad, int tid) {
  for (int i = tid; i < Npad; i += NTHREADS) dst[i] = (i < Nreal) ? b[i] : 0.0f;
}

// ---------------------------------------------------------------------------
__global__ __launch_bounds__(NTHREADS, 1)
void neural_ode_rk4_kernel(const float* __restrict__ x,
                           const float* enc_w1, const float* enc_b1,
                           const float* enc_w2, const float* enc_b2,
                           const float* ode_w1, const float* ode_b1,
                           const float* ode_w2, const float* ode_b2,
                           const float* ode_w3, const float* ode_b3,
                           const float* dec_w1, const float* dec_b1,
                           const float* dec_w2, const float* dec_b2,
                           float* __restrict__ out) {
  extern __shared__ char smem[];
  bf16*  WtE1 = (bf16*)(smem + OFF_WTE1);
  bf16*  WtE2 = (bf16*)(smem + OFF_WTE2);
  bf16*  Wt1  = (bf16*)(smem + OFF_WT1);
  bf16*  Wt2  = (bf16*)(smem + OFF_WT2);
  bf16*  Wt3  = (bf16*)(smem + OFF_WT3);
  bf16*  WtD1 = (bf16*)(smem + OFF_WTD1);
  bf16*  WtD2 = (bf16*)(smem + OFF_WTD2);
  float* bE1  = (float*)(smem + OFF_BE1);
  float* bE2  = (float*)(smem + OFF_BE2);
  float* b1   = (float*)(smem + OFF_B1);
  float* b2   = (float*)(smem + OFF_B2);
  float* b3   = (float*)(smem + OFF_B3);
  float* bD1  = (float*)(smem + OFF_BD1);
  float* bD2  = (float*)(smem + OFF_BD2);
  bf16*  zb   = (bf16*)(smem + OFF_ZB);
  bf16*  zpert= (bf16*)(smem + OFF_ZPERT);
  bf16*  ubuf = (bf16*)(smem + OFF_UBUF);
  bf16*  hA   = (bf16*)(smem + OFF_HA);
  bf16*  hB   = (bf16*)(smem + OFF_HB);

  const int tid  = threadIdx.x;
  const int lane = tid & 31;
  const int w    = tid >> 5;          // wave id 0..7
  const int B0   = blockIdx.x * 16;   // batch tile base row

  // Per-wave-owned z slice (waves 0..3): col n = w*16+(lane&15),
  // rows mb..mb+7 with mb = (lane>>4)*8 — identical to WMMA C layout.
  const int zn  = w * 16 + (lane & 15);
  const int zmb = (lane >> 4) << 3;

  // ---- one-time weight conversion into LDS ----
  load_wt(enc_w1, INPUT,  HIDDEN, WtE1, 32,  128, tid);
  load_wt(enc_w2, HIDDEN, LATENT, WtE2, 128, 64,  tid);
  load_wt(ode_w1, LATENT + INPUT, HIDDEN, Wt1, 96,  128, tid);
  load_wt(ode_w2, HIDDEN, HIDDEN, Wt2, 128, 128, tid);
  load_wt(ode_w3, HIDDEN, LATENT, Wt3, 128, 64,  tid);
  load_wt(dec_w1, LATENT, HIDDEN, WtD1, 64, 128, tid);
  load_wt(dec_w2, HIDDEN, OUTPUT, WtD2, 128, 32, tid);
  load_bias(enc_b1, HIDDEN, bE1, 128, tid);
  load_bias(enc_b2, LATENT, bE2, 64,  tid);
  load_bias(ode_b1, HIDDEN, b1,  128, tid);
  load_bias(ode_b2, HIDDEN, b2,  128, tid);
  load_bias(ode_b3, LATENT, b3,  64,  tid);
  load_bias(dec_b1, HIDDEN, bD1, 128, tid);
  load_bias(dec_b2, OUTPUT, bD2, 32,  tid);
  __syncthreads();

  // ---- encoder on t=0: z0 = tanh(x0 @ We1 + be1) @ We2 + be2 ----
  for (int i = tid; i < 16 * 32; i += NTHREADS) {
    const int m = i >> 5, c = i & 31;
    const float v = (c < INPUT) ? x[((size_t)(B0 + m) * SEQ + 0) * INPUT + c] : 0.0f;
    ubuf[i] = (bf16)v;
  }
  __syncthreads();
  {
    const int nb = w * 16;
    v8f acc = wmma_bf16(load_a32(ubuf, 32, 0, lane),
                        load_b32(WtE1, 32, nb, 0, lane), vzero());
    store_act_tanh(hA, HIDDEN, nb, lane, acc, bE1);
  }
  __syncthreads();

  v8f zcur = vzero();   // register-resident z state (waves 0..3)
  v8f zacc = vzero();
  if (w < 4) {
    const int nb = w * 16;
    v8f acc = vzero();
#pragma unroll
    for (int kb = 0; kb < 128; kb += 32)
      acc = wmma_bf16(load_a32(hA, HIDDEN, kb, lane),
                      load_b32(WtE2, 128, nb, kb, lane), acc);
    const float bn = bE2[zn];
#pragma unroll
    for (int r = 0; r < 8; ++r) {
      zcur[r] = acc[r] + bn;
      zb[(zmb + r) * 64 + zn] = (bf16)zcur[r];
    }
  }
  __syncthreads();

  // ---- scan over time ----
  for (int t = 0; t < SEQ; ++t) {
    // stage u(t) into LDS (zero-padded 21 -> 32)
    for (int i = tid; i < 16 * 32; i += NTHREADS) {
      const int m = i >> 5, c = i & 31;
      const float v = (c < INPUT) ? x[((size_t)(B0 + m) * SEQ + t) * INPUT + c] : 0.0f;
      ubuf[i] = (bf16)v;
    }
    if (t + 1 < SEQ && tid < 16)  // global_prefetch_b8 for next step's rows
      __builtin_prefetch(&x[((size_t)(B0 + tid) * SEQ + (t + 1)) * INPUT], 0, 0);
    __syncthreads();

    // ---- decode(z): out_t = sigmoid(tanh(z @ Wd1 + bd1) @ Wd2 + bd2) ----
    {
      const int nb = w * 16;
      v8f acc = vzero();
      acc = wmma_bf16(load_a32(zb, 64, 0,  lane), load_b32(WtD1, 64, nb, 0,  lane), acc);
      acc = wmma_bf16(load_a32(zb, 64, 32, lane), load_b32(WtD1, 64, nb, 32, lane), acc);
      store_act_tanh(hA, HIDDEN, nb, lane, acc, bD1);
    }
    __syncthreads();
    if (w < 2) {
      const int nb = w * 16;
      v8f acc = vzero();
#pragma unroll
      for (int kb = 0; kb < 128; kb += 32)
        acc = wmma_bf16(load_a32(hA, HIDDEN, kb, lane),
                        load_b32(WtD2, 128, nb, kb, lane), acc);
      const int n  = nb + (lane & 15);
      const int mb = (lane >> 4) << 3;
      if (n < OUTPUT) {
        const float bn = bD2[n];
#pragma unroll
        for (int r = 0; r < 8; ++r)
          out[((size_t)(B0 + mb + r) * SEQ + t) * OUTPUT + n] =
              fast_sigmoid(acc[r] + bn);
      }
    }
    __syncthreads();

    // ---- RK4: z <- z + dt/6 (k1 + 2k2 + 2k3 + k4) ----
    v8f k = ode_f(zb, ubuf, Wt1, b1, Wt2, b2, Wt3, b3, hA, hB, w, lane);
    if (w < 4) {
#pragma unroll
      for (int r = 0; r < 8; ++r) {
        zacc[r] = zcur[r] + (DT / 6.0f) * k[r];
        zpert[(zmb + r) * 64 + zn] = (bf16)(zcur[r] + 0.5f * DT * k[r]);
      }
    }
    __syncthreads();
    k = ode_f(zpert, ubuf, Wt1, b1, Wt2, b2, Wt3, b3, hA, hB, w, lane);
    if (w < 4) {
#pragma unroll
      for (int r = 0; r < 8; ++r) {
        zacc[r] += (DT / 3.0f) * k[r];
        zpert[(zmb + r) * 64 + zn] = (bf16)(zcur[r] + 0.5f * DT * k[r]);
      }
    }
    __syncthreads();
    k = ode_f(zpert, ubuf, Wt1, b1, Wt2, b2, Wt3, b3, hA, hB, w, lane);
    if (w < 4) {
#pragma unroll
      for (int r = 0; r < 8; ++r) {
        zacc[r] += (DT / 3.0f) * k[r];
        zpert[(zmb + r) * 64 + zn] = (bf16)(zcur[r] + DT * k[r]);
      }
    }
    __syncthreads();
    k = ode_f(zpert, ubuf, Wt1, b1, Wt2, b2, Wt3, b3, hA, hB, w, lane);
    if (w < 4) {
#pragma unroll
      for (int r = 0; r < 8; ++r) {
        zcur[r] = zacc[r] + (DT / 6.0f) * k[r];
        zb[(zmb + r) * 64 + zn] = (bf16)zcur[r];
      }
    }
    __syncthreads();
  }
}

// ---------------------------------------------------------------------------
extern "C" void kernel_launch(void* const* d_in, const int* in_sizes, int n_in,
                              void* d_out, int out_size, void* d_ws, size_t ws_size,
                              hipStream_t stream) {
  (void)in_sizes; (void)n_in; (void)out_size; (void)d_ws; (void)ws_size;
  const float* x      = (const float*)d_in[0];
  const float* enc_w1 = (const float*)d_in[1];
  const float* enc_b1 = (const float*)d_in[2];
  const float* enc_w2 = (const float*)d_in[3];
  const float* enc_b2 = (const float*)d_in[4];
  const float* ode_w1 = (const float*)d_in[5];
  const float* ode_b1 = (const float*)d_in[6];
  const float* ode_w2 = (const float*)d_in[7];
  const float* ode_b2 = (const float*)d_in[8];
  const float* ode_w3 = (const float*)d_in[9];
  const float* ode_b3 = (const float*)d_in[10];
  const float* dec_w1 = (const float*)d_in[11];
  const float* dec_b1 = (const float*)d_in[12];
  const float* dec_w2 = (const float*)d_in[13];
  const float* dec_b2 = (const float*)d_in[14];
  float* out = (float*)d_out;

  static bool attr_set = false;
  if (!attr_set) {  // allow >64KB dynamic LDS (320KB WGP on gfx1250)
    (void)hipFuncSetAttribute((const void*)neural_ode_rk4_kernel,
                              hipFuncAttributeMaxDynamicSharedMemorySize,
                              SMEM_BYTES);
    attr_set = true;
  }

  dim3 grid(BATCH / 16);
  dim3 block(NTHREADS);
  neural_ode_rk4_kernel<<<grid, block, SMEM_BYTES, stream>>>(
      x, enc_w1, enc_b1, enc_w2, enc_b2,
      ode_w1, ode_b1, ode_w2, ode_b2, ode_w3, ode_b3,
      dec_w1, dec_b1, dec_w2, dec_b2, out);
}
